// DEN_39771397161306
// MI455X (gfx1250) — compile-verified, matrix operand order checked by reading, not run on hardware
//
#include <hip/hip_runtime.h>
#include <hip/hip_bf16.h>
#include <math.h>

// ---------------- CDNA5 WMMA types ----------------
typedef __bf16 bf16;
typedef __attribute__((ext_vector_type(16))) __bf16 v16bf;
typedef __attribute__((ext_vector_type(8)))  float  v8f;

#define LD  200          // LDS row stride for activation tiles (even, non-pow2: dodges bank conflicts)
#define NW  8            // waves per block (256 threads, wave32)

union Frag { v16bf v; unsigned u[8]; bf16 e[16]; };

__device__ __forceinline__ v8f v8f_zero() {
    v8f z = {0.f,0.f,0.f,0.f,0.f,0.f,0.f,0.f};
    return z;
}

// K-offset of packed bf16 pair v (0..7) for lane-half sel, per CDNA5 16-bit A/B layout:
// lanes 0-15: VGPR0-3 -> K=0..7, VGPR4-7 -> K=16..23 ; lanes 16-31: +8
__device__ __forceinline__ int kpair(int v, int sel) {
    return ((v < 4) ? (2 * v) : (16 + 2 * (v - 4))) + 8 * sel;
}

// A fragment (16x32 bf16, M x K) from row-major LDS [m][k]; packed 32-bit DS loads.
// Also used for B fragments when the weight is stored K-transposed ([n][k]).
__device__ __forceinline__ v16bf load_a(const bf16* base, int ld, int m0, int k0) {
    const int lane = threadIdx.x & 31;
    const int m = m0 + (lane & 15), sel = lane >> 4;
    Frag f;
    const unsigned* row = (const unsigned*)(base + m * ld + k0);
#pragma unroll
    for (int v = 0; v < 8; ++v) f.u[v] = row[kpair(v, sel) >> 1];
    return f.v;
}

// A fragment where logical A[m][k] = base[k*ld + m]  (transposed read)
__device__ __forceinline__ v16bf load_a_t(const bf16* base, int ld, int m0, int k0) {
    const int lane = threadIdx.x & 31;
    const int m = m0 + (lane & 15), sel = lane >> 4;
    Frag f;
#pragma unroll
    for (int e = 0; e < 16; ++e) {
        int k = k0 + kpair(e >> 1, sel) + (e & 1);
        f.e[e] = base[k * ld + m];
    }
    return f.v;
}

// B fragment (32x16 bf16, K x N) from row-major LDS [k][n]
__device__ __forceinline__ v16bf load_b_lds(const bf16* base, int ld, int k0, int n0) {
    const int lane = threadIdx.x & 31;
    const int n = n0 + (lane & 15), sel = lane >> 4;
    Frag f;
#pragma unroll
    for (int e = 0; e < 16; ++e) {
        int k = k0 + kpair(e >> 1, sel) + (e & 1);
        f.e[e] = base[k * ld + n];
    }
    return f.v;
}

// B fragment from global f32 weights [k][n] (fallback path; rows >= kmax read as 0)
__device__ __forceinline__ v16bf load_b_g(const float* W, int ldn, int k0, int n0, int kmax) {
    const int lane = threadIdx.x & 31;
    const int n = n0 + (lane & 15), sel = lane >> 4;
    Frag f;
#pragma unroll
    for (int e = 0; e < 16; ++e) {
        int k = k0 + kpair(e >> 1, sel) + (e & 1);
        f.e[e] = (k < kmax) ? (bf16)W[k * ldn + n] : (bf16)0.f;
    }
    return f.v;
}

__device__ __forceinline__ v8f wmma_bf(v16bf a, v16bf b, v8f c) {
    // (neg_a, A, neg_b, B, c_mod, C, reuse_a, reuse_b)
    return __builtin_amdgcn_wmma_f32_16x16x32_bf16(false, a, false, b, (short)0, c, false, false);
}

__device__ __forceinline__ float gelu_f(float x) {
    return 0.5f * x * (1.f + erff(x * 0.70710678118654752f));
}

// ---- CDNA5 async DMA: global -> LDS, ASYNCcnt-tracked (no VGPR round trip) ----
// LDS generic pointers carry the LDS byte offset in the low 32 bits (aperture rule),
// so truncation gives the wave-relative LDS address the instruction wants.
__device__ __forceinline__ void stage_async(const void* gsrc, void* ldst, int bytes) {
    uint32_t l0 = (uint32_t)(uintptr_t)ldst;
    uint64_t g0 = (uint64_t)(uintptr_t)gsrc;
    for (int off = (int)threadIdx.x * 16; off < bytes; off += 256 * 16) {
        uint32_t la = l0 + (uint32_t)off;
        uint64_t ga = g0 + (uint64_t)off;
        asm volatile("global_load_async_to_lds_b128 %0, %1, off"
                     :: "v"(la), "v"(ga) : "memory");
    }
}
__device__ __forceinline__ void wait_async() {
    asm volatile("s_wait_asynccnt 0" ::: "memory");
}

// ---------------- parameter block ----------------
struct LayerP {
    const float *K, *WK, *WQ, *WV, *bK, *bQ, *bV, *delta_t;
    const float *ff_W1, *ff_W2, *ff_b1, *ff_b2;
    const float *lu_W1, *lu_W2, *lu_b1, *lu_b2;
    const float *n1_b, *n1_g, *n2_b, *n2_g;
    const float *tm_W1, *tm_W2, *tm_b, *tm_b1, *tm_b2, *tm_g;
};
struct DenArgs {
    const float *x, *patch_W, *patch_b, *pos, *head_W, *head_b;
    LayerP L[6];
    const bf16* wsb;     // bf16 transposed weights in d_ws (staged mode)
    float* out;
};

// bf16 weight workspace layout (element offsets), all matrices stored [n][k] (K-transposed)
#define WS_PATCH     0            // [192][64]  (K=48 zero-padded to 64)
#define WS_LAYER0    12288
#define WS_LSIZE     266240
#define WSO_LU1      0            // [192][192]
#define WSO_LU2      36864
#define WSO_WQ       73728
#define WSO_WK       110592
#define WSO_WV       147456
#define WSO_FF1      184320
#define WSO_FF2      221184
#define WSO_TM1      258048       // [64][64]
#define WSO_TM2      262144
#define WS_TOTAL_ELEMS (WS_LAYER0 + 6 * WS_LSIZE)     // 1,609,728 bf16 = 3,219,456 B

// ---------------- LDS layout (bytes) ----------------
#define OFF_X     0                       // bf16 [64][LD]   activations
#define OFF_S     25600                   // f32  [64][LD]   residual accumulator
#define OFF_SCR   76800                   // bf16 [192][LD]  scratch (hidden / XN / Qp,Kp,Vp)
#define OFF_SCR2  153600                  // bf16 [192][LD]  scratch (weights / token hidden / KV)
#define OFF_KP    230400                  // bf16 [64][72]   Kpos / patch staging / tm weights
#define OFF_RS    239616                  // f32  [64]
#define OFF_KSUM  239872                  // f32  [192]
#define OFF_NRM   240640                  // f32  [64]
#define SMEM_BYTES 240896                 // ~235 KB of the 320 KB/WGP LDS

// two-GEMM MLP (X@W1 -> gelu -> @W2), accumulates into S.
// STAGED: weights async-DMA'd into WL (LDS) as [n][k] bf16, B frags are packed loads.
template<bool STAGED>
__device__ __forceinline__ void mlp192(const bf16* Xin, bf16* H, float* Sacc,
                                       const float* W1f, const bf16* W1t, const float* b1,
                                       const float* W2f, const bf16* W2t, const float* b2,
                                       bf16* WL) {
    const int tid = threadIdx.x, wave = tid >> 5, lane = tid & 31;
    const int nloc = lane & 15, sel = lane >> 4;
    if constexpr (STAGED) { stage_async(W1t, WL, 192 * 192 * 2); wait_async(); }
    __syncthreads();
    for (int t = wave; t < 48; t += NW) {
        int mt = t / 12, nt = t % 12;
        v8f acc = v8f_zero();
#pragma unroll
        for (int ks = 0; ks < 6; ++ks) {
            v16bf bf;
            if constexpr (STAGED) bf = load_a(WL, 192, nt * 16, ks * 32);
            else                  bf = load_b_g(W1f, 192, ks * 32, nt * 16, 192);
            acc = wmma_bf(load_a(Xin, LD, mt * 16, ks * 32), bf, acc);
        }
        int n = nt * 16 + nloc;
#pragma unroll
        for (int i = 0; i < 8; ++i)
            H[(mt * 16 + i + 8 * sel) * LD + n] = (bf16)gelu_f(acc[i] + b1[n]);
    }
    __syncthreads();
    if constexpr (STAGED) { stage_async(W2t, WL, 192 * 192 * 2); wait_async(); }
    __syncthreads();
    for (int t = wave; t < 48; t += NW) {
        int mt = t / 12, nt = t % 12;
        v8f acc = v8f_zero();
#pragma unroll
        for (int ks = 0; ks < 6; ++ks) {
            v16bf bf;
            if constexpr (STAGED) bf = load_a(WL, 192, nt * 16, ks * 32);
            else                  bf = load_b_g(W2f, 192, ks * 32, nt * 16, 192);
            acc = wmma_bf(load_a(H, LD, mt * 16, ks * 32), bf, acc);
        }
        int n = nt * 16 + nloc;
#pragma unroll
        for (int i = 0; i < 8; ++i) {
            int m = mt * 16 + i + 8 * sel;
            Sacc[m * LD + n] += acc[i] + b2[n];
        }
    }
    __syncthreads();
}

// LayerNorm over D=192 of f32 rows in S; writes bf16 X and f32 S in place.
__device__ __forceinline__ void ln_f32(float* Ssrc, bf16* Xdst,
                                       const float* g, const float* bb) {
    int row = threadIdx.x >> 2, r4 = threadIdx.x & 3;
    float* sr = Ssrc + row * LD;
    float s = 0.f, s2 = 0.f;
    for (int j = r4; j < 192; j += 4) { float v = sr[j]; s += v; s2 += v * v; }
    s  += __shfl_xor(s, 1);  s  += __shfl_xor(s, 2);
    s2 += __shfl_xor(s2, 1); s2 += __shfl_xor(s2, 2);
    float mean = s * (1.f / 192.f);
    float var  = fmaxf(s2 * (1.f / 192.f) - mean * mean, 0.f);
    float inv  = rsqrtf(var + 1e-5f);
    for (int j = r4; j < 192; j += 4) {
        float v = (sr[j] - mean) * inv * g[j] + bb[j];
        Xdst[row * LD + j] = (bf16)v;
        sr[j] = v;
    }
}

template<bool STAGED>
__global__ __launch_bounds__(256) void den_kernel(DenArgs A) {
    extern __shared__ char smem[];
    bf16*  X    = (bf16*)(smem + OFF_X);
    float* S    = (float*)(smem + OFF_S);
    bf16*  SCR  = (bf16*)(smem + OFF_SCR);
    bf16*  SCR2 = (bf16*)(smem + OFF_SCR2);
    bf16*  KP   = (bf16*)(smem + OFF_KP);
    float* RS   = (float*)(smem + OFF_RS);
    float* KSUM = (float*)(smem + OFF_KSUM);
    float* NRM  = (float*)(smem + OFF_NRM);

    const int tid  = threadIdx.x;
    const int wave = tid >> 5;
    const int lane = tid & 31;
    const int nloc = lane & 15;
    const int sel  = lane >> 4;
    const int b    = blockIdx.x;

    // ================= patch embed =================
    {
        if constexpr (STAGED) stage_async(A.wsb + WS_PATCH, SCR2, 192 * 64 * 2);
        const float* xg = A.x + (size_t)b * 3 * 32 * 32;
        for (int e2 = tid; e2 < 64 * 64; e2 += 256) {    // stage patches [64][48] zero-padded to K=64
            int t = e2 >> 6, col = e2 & 63;
            float v = 0.f;
            if (col < 48) {
                int c = col >> 4, py = (col >> 2) & 3, px = col & 3;
                int hp = t >> 3, wp = t & 7;
                v = xg[(c * 32 + hp * 4 + py) * 32 + wp * 4 + px];
            }
            KP[t * 72 + col] = (bf16)v;
        }
        if constexpr (STAGED) wait_async();
        __syncthreads();
        for (int t = wave; t < 48; t += NW) {            // [64,48]@[48,192]
            int mt = t / 12, nt = t % 12;
            v8f acc = v8f_zero();
#pragma unroll
            for (int ks = 0; ks < 2; ++ks) {
                v16bf bf;
                if constexpr (STAGED) bf = load_a(SCR2, 64, nt * 16, ks * 32);
                else                  bf = load_b_g(A.patch_W, 192, ks * 32, nt * 16, 48);
                acc = wmma_bf(load_a(KP, 72, mt * 16, ks * 32), bf, acc);
            }
            int n = nt * 16 + nloc;
#pragma unroll
            for (int i = 0; i < 8; ++i) {
                int m = mt * 16 + i + 8 * sel;
                float v = acc[i] + A.patch_b[n] + A.pos[m * 192 + n];
                X[m * LD + n] = (bf16)v;
                S[m * LD + n] = v;
            }
        }
        __syncthreads();
    }

    // ================= layers =================
    for (int l = 0; l < 6; ++l) {
        const LayerP& P = A.L[l];
        const bf16* WB = A.wsb + WS_LAYER0 + l * WS_LSIZE;

        // ---- diffusion: S += dt*(softplus(K)@x - rowsum*x) ----
        if (tid < 64) {
            float rs = 0.f;
            for (int j = 0; j < 64; ++j) {
                float kv = P.K[tid * 64 + j];
                float sp = (kv > 20.f) ? kv : log1pf(expf(kv));
                KP[tid * 72 + j] = (bf16)sp;
                rs += sp;
            }
            RS[tid] = rs;
        }
        __syncthreads();
        {
            float dt = P.delta_t[0];
            for (int t = wave; t < 48; t += NW) {        // [64,64]@[64,192]
                int mt = t / 12, nt = t % 12;
                v8f acc = v8f_zero();
#pragma unroll
                for (int ks = 0; ks < 2; ++ks)
                    acc = wmma_bf(load_a(KP, 72, mt * 16, ks * 32),
                                  load_b_lds(X, LD, ks * 32, nt * 16), acc);
                int n = nt * 16 + nloc;
#pragma unroll
                for (int i = 0; i < 8; ++i) {
                    int m = mt * 16 + i + 8 * sel;
                    S[m * LD + n] += dt * (acc[i] - RS[m] * (float)X[m * LD + n]);
                }
            }
        }
        __syncthreads();

        // ---- local per-token MLP ----
        mlp192<STAGED>(X, SCR, S, P.lu_W1, WB + WSO_LU1, P.lu_b1,
                                  P.lu_W2, WB + WSO_LU2, P.lu_b2, SCR2);

        // ---- token mixing: LN -> transpose -> MLP over tokens ----
        if constexpr (STAGED) stage_async(WB + WSO_TM1, KP, 64 * 64 * 2);  // overlaps with LN
        {   // LN(X) -> XN in SCR rows 0..63
            int row = tid >> 2, r4 = tid & 3;
            const bf16* sr = X + row * LD;
            float s = 0.f, s2 = 0.f;
            for (int j = r4; j < 192; j += 4) { float v = (float)sr[j]; s += v; s2 += v * v; }
            s  += __shfl_xor(s, 1);  s  += __shfl_xor(s, 2);
            s2 += __shfl_xor(s2, 1); s2 += __shfl_xor(s2, 2);
            float mean = s * (1.f / 192.f);
            float inv  = rsqrtf(fmaxf(s2 * (1.f / 192.f) - mean * mean, 0.f) + 1e-5f);
            for (int j = r4; j < 192; j += 4) {
                float v = ((float)sr[j] - mean) * inv * P.tm_g[j] + P.tm_b[j];
                SCR[row * LD + j] = (bf16)v;
            }
        }
        if constexpr (STAGED) wait_async();
        __syncthreads();
        for (int t = wave; t < 48; t += NW) {            // Xn^T [192,64] @ tm_W1 [64,64]
            int mt = t / 4, nt = t % 4;
            v8f acc = v8f_zero();
#pragma unroll
            for (int ks = 0; ks < 2; ++ks) {
                v16bf bf;
                if constexpr (STAGED) bf = load_a(KP, 64, nt * 16, ks * 32);
                else                  bf = load_b_g(P.tm_W1, 64, ks * 32, nt * 16, 64);
                acc = wmma_bf(load_a_t(SCR, LD, mt * 16, ks * 32), bf, acc);
            }
            int n = nt * 16 + nloc;
#pragma unroll
            for (int i = 0; i < 8; ++i)
                SCR2[(mt * 16 + i + 8 * sel) * LD + n] = (bf16)gelu_f(acc[i] + P.tm_b1[n]);
        }
        __syncthreads();
        if constexpr (STAGED) { stage_async(WB + WSO_TM2, KP, 64 * 64 * 2); wait_async(); }
        __syncthreads();
        for (int t = wave; t < 48; t += NW) {            // H [192,64] @ tm_W2 [64,64], store transposed into S
            int mt = t / 4, nt = t % 4;
            v8f acc = v8f_zero();
#pragma unroll
            for (int ks = 0; ks < 2; ++ks) {
                v16bf bf;
                if constexpr (STAGED) bf = load_a(KP, 64, nt * 16, ks * 32);
                else                  bf = load_b_g(P.tm_W2, 64, ks * 32, nt * 16, 64);
                acc = wmma_bf(load_a(SCR2, LD, mt * 16, ks * 32), bf, acc);
            }
            int n = nt * 16 + nloc;                      // n = token
#pragma unroll
            for (int i = 0; i < 8; ++i) {
                int m = mt * 16 + i + 8 * sel;           // m = embed dim
                S[n * LD + m] += acc[i] + P.tm_b2[n];
            }
        }
        __syncthreads();

        // ---- linear attention ----
        for (int qkv = 0; qkv < 3; ++qkv) {              // Q,K,V projections (+ elu+1 on Q,K)
            const float* Wf = (qkv == 0) ? P.WQ : (qkv == 1) ? P.WK : P.WV;
            const bf16*  Wt = WB + ((qkv == 0) ? WSO_WQ : (qkv == 1) ? WSO_WK : WSO_WV);
            const float* bb = (qkv == 0) ? P.bQ : (qkv == 1) ? P.bK : P.bV;
            bf16* dst = SCR + qkv * 64 * LD;
            if constexpr (STAGED) { stage_async(Wt, SCR2, 192 * 192 * 2); wait_async(); }
            __syncthreads();
            for (int t = wave; t < 48; t += NW) {
                int mt = t / 12, nt = t % 12;
                v8f acc = v8f_zero();
#pragma unroll
                for (int ks = 0; ks < 6; ++ks) {
                    v16bf bf;
                    if constexpr (STAGED) bf = load_a(SCR2, 192, nt * 16, ks * 32);
                    else                  bf = load_b_g(Wf, 192, ks * 32, nt * 16, 192);
                    acc = wmma_bf(load_a(X, LD, mt * 16, ks * 32), bf, acc);
                }
                int n = nt * 16 + nloc;
#pragma unroll
                for (int i = 0; i < 8; ++i) {
                    float v = acc[i] + bb[n];
                    if (qkv < 2) v = (v > 0.f) ? (v + 1.f) : expf(v);  // elu(v)+1
                    dst[(mt * 16 + i + 8 * sel) * LD + n] = (bf16)v;
                }
            }
            __syncthreads();
        }
        if (tid < 192) {                                 // KSUM[d] = sum_t Kp[t][d]
            float s = 0.f;
            for (int tt = 0; tt < 64; ++tt) s += (float)SCR[(64 + tt) * LD + tid];
            KSUM[tid] = s;
        }
        __syncthreads();
        if (tid < 64) {                                  // NRM[t] = Qp[t]·KSUM
            float s = 0.f;
            for (int d = 0; d < 192; ++d) s += (float)SCR[tid * LD + d] * KSUM[d];
            NRM[tid] = s;
        }
        for (int t = wave; t < 144; t += NW) {           // KV = Kp^T @ Vp  [192,192]
            int mt = t / 12, nt = t % 12;
            v8f acc = v8f_zero();
#pragma unroll
            for (int ks = 0; ks < 2; ++ks)
                acc = wmma_bf(load_a_t(SCR + 64 * LD, LD, mt * 16, ks * 32),
                              load_b_lds(SCR + 128 * LD, LD, ks * 32, nt * 16), acc);
            int n = nt * 16 + nloc;
#pragma unroll
            for (int i = 0; i < 8; ++i)
                SCR2[(mt * 16 + i + 8 * sel) * LD + n] = (bf16)acc[i];
        }
        __syncthreads();
        for (int t = wave; t < 48; t += NW) {            // num = Qp @ KV ; S += num/(nrm+eps)
            int mt = t / 12, nt = t % 12;
            v8f acc = v8f_zero();
#pragma unroll
            for (int ks = 0; ks < 6; ++ks)
                acc = wmma_bf(load_a(SCR, LD, mt * 16, ks * 32),
                              load_b_lds(SCR2, LD, ks * 32, nt * 16), acc);
            int n = nt * 16 + nloc;
#pragma unroll
            for (int i = 0; i < 8; ++i) {
                int m = mt * 16 + i + 8 * sel;
                S[m * LD + n] += acc[i] / (NRM[m] + 1e-6f);
            }
        }
        __syncthreads();

        // ---- LN1 -> FF -> LN2 ----
        ln_f32(S, X, P.n1_g, P.n1_b);
        __syncthreads();
        mlp192<STAGED>(X, SCR, S, P.ff_W1, WB + WSO_FF1, P.ff_b1,
                                  P.ff_W2, WB + WSO_FF2, P.ff_b2, SCR2);
        ln_f32(S, X, P.n2_g, P.n2_b);
        __syncthreads();
    }

    // ================= head: mean-pool + [192,10] ================
    if (tid < 192) {
        float s = 0.f;
        for (int tt = 0; tt < 64; ++tt) s += S[tt * LD + tid];
        KSUM[tid] = s * (1.f / 64.f);
    }
    __syncthreads();
    if (tid < 10) {
        float s = A.head_b[tid];
        for (int d = 0; d < 192; ++d) s += KSUM[d] * A.head_W[d * 10 + tid];
        A.out[(size_t)b * 10 + tid] = s;
    }
}

// --------- weight convert/transpose pre-pass: f32 [K][N] -> bf16 [N][Kpad] ---------
struct CvtEnt { const float* src; int dstOff, K, N, Kpad; };
struct CvtArgs { CvtEnt e[56]; bf16* dst; };

__global__ __launch_bounds__(256) void cvt_kernel(CvtArgs C) {
    const CvtEnt E = C.e[blockIdx.x];
    bf16* d = C.dst + E.dstOff;
    int tot = E.N * E.Kpad;
    for (int i = threadIdx.x; i < tot; i += 256) {
        int n = i / E.Kpad, k = i - n * E.Kpad;
        d[i] = (k < E.K) ? (bf16)E.src[(size_t)k * E.N + n] : (bf16)0.f;
    }
}

// ---------------- host launcher ----------------
extern "C" void kernel_launch(void* const* d_in, const int* in_sizes, int n_in,
                              void* d_out, int out_size, void* d_ws, size_t ws_size,
                              hipStream_t stream) {
    (void)out_size;
    const float* const* in = (const float* const*)d_in;
    DenArgs A;
    int xIdx = 0;
    if (n_in >= 1 && in_sizes[0] > 100000) {
        // insertion-order flattening: x first, then params in declaration order
        A.x = in[0]; xIdx = 0;
        A.patch_W = in[1]; A.patch_b = in[2]; A.pos = in[3];
        A.head_W  = in[4]; A.head_b  = in[5];
        int idx = 6;
        for (int l = 0; l < 6; ++l, idx += 26) {
            LayerP& L = A.L[l];
            L.K = in[idx + 0];  L.delta_t = in[idx + 1];
            L.lu_W1 = in[idx + 2];  L.lu_b1 = in[idx + 3];
            L.lu_W2 = in[idx + 4];  L.lu_b2 = in[idx + 5];
            L.tm_g  = in[idx + 6];  L.tm_b  = in[idx + 7];
            L.tm_W1 = in[idx + 8];  L.tm_b1 = in[idx + 9];
            L.tm_W2 = in[idx + 10]; L.tm_b2 = in[idx + 11];
            L.WQ = in[idx + 12]; L.bQ = in[idx + 13];
            L.WK = in[idx + 14]; L.bK = in[idx + 15];
            L.WV = in[idx + 16]; L.bV = in[idx + 17];
            L.ff_W1 = in[idx + 18]; L.ff_b1 = in[idx + 19];
            L.ff_W2 = in[idx + 20]; L.ff_b2 = in[idx + 21];
            L.n1_g = in[idx + 22]; L.n1_b = in[idx + 23];
            L.n2_g = in[idx + 24]; L.n2_b = in[idx + 25];
        }
    } else {
        // JAX pytree order: dict keys sorted (ASCII); params before x
        A.head_W = in[0]; A.head_b = in[1];
        int idx = 2;
        for (int l = 0; l < 6; ++l, idx += 26) {
            LayerP& L = A.L[l];
            L.K = in[idx + 0];
            L.WK = in[idx + 1]; L.WQ = in[idx + 2]; L.WV = in[idx + 3];
            L.bK = in[idx + 4]; L.bQ = in[idx + 5]; L.bV = in[idx + 6];
            L.delta_t = in[idx + 7];
            L.ff_W1 = in[idx + 8];  L.ff_W2 = in[idx + 9];
            L.ff_b1 = in[idx + 10]; L.ff_b2 = in[idx + 11];
            L.lu_W1 = in[idx + 12]; L.lu_W2 = in[idx + 13];
            L.lu_b1 = in[idx + 14]; L.lu_b2 = in[idx + 15];
            L.n1_b = in[idx + 16]; L.n1_g = in[idx + 17];
            L.n2_b = in[idx + 18]; L.n2_g = in[idx + 19];
            L.tm_W1 = in[idx + 20]; L.tm_W2 = in[idx + 21];
            L.tm_b  = in[idx + 22]; L.tm_b1 = in[idx + 23];
            L.tm_b2 = in[idx + 24]; L.tm_g  = in[idx + 25];
        }
        A.patch_W = in[idx + 0]; A.patch_b = in[idx + 1]; A.pos = in[idx + 2];
        A.x = in[idx + 3]; xIdx = idx + 3;
    }
    A.out = (float*)d_out;
    A.wsb = (const bf16*)d_ws;
    int nB = in_sizes[xIdx] / (3 * 32 * 32);
    if (nB < 1) nB = 1;

    bool staged = (d_ws != nullptr) && (ws_size >= (size_t)WS_TOTAL_ELEMS * 2);
    if (staged) {
        CvtArgs C;
        C.dst = (bf16*)d_ws;
        int ne = 0;
        C.e[ne++] = {A.patch_W, WS_PATCH, 48, 192, 64};
        for (int l = 0; l < 6; ++l) {
            int base = WS_LAYER0 + l * WS_LSIZE;
            C.e[ne++] = {A.L[l].lu_W1, base + WSO_LU1, 192, 192, 192};
            C.e[ne++] = {A.L[l].lu_W2, base + WSO_LU2, 192, 192, 192};
            C.e[ne++] = {A.L[l].WQ,    base + WSO_WQ,  192, 192, 192};
            C.e[ne++] = {A.L[l].WK,    base + WSO_WK,  192, 192, 192};
            C.e[ne++] = {A.L[l].WV,    base + WSO_WV,  192, 192, 192};
            C.e[ne++] = {A.L[l].ff_W1, base + WSO_FF1, 192, 192, 192};
            C.e[ne++] = {A.L[l].ff_W2, base + WSO_FF2, 192, 192, 192};
            C.e[ne++] = {A.L[l].tm_W1, base + WSO_TM1, 64, 64, 64};
            C.e[ne++] = {A.L[l].tm_W2, base + WSO_TM2, 64, 64, 64};
        }
        for (int i = ne; i < 56; ++i) C.e[i] = {nullptr, 0, 0, 0, 1};
        cvt_kernel<<<dim3(ne), dim3(256), 0, stream>>>(C);
        hipFuncSetAttribute(reinterpret_cast<const void*>(&den_kernel<true>),
                            hipFuncAttributeMaxDynamicSharedMemorySize, SMEM_BYTES);
        den_kernel<true><<<dim3(nB), dim3(256), SMEM_BYTES, stream>>>(A);
    } else {
        hipFuncSetAttribute(reinterpret_cast<const void*>(&den_kernel<false>),
                            hipFuncAttributeMaxDynamicSharedMemorySize, SMEM_BYTES);
        den_kernel<false><<<dim3(nB), dim3(256), SMEM_BYTES, stream>>>(A);
    }
}
